// VariableLengthEncoder_17085379904340
// MI455X (gfx1250) — matrix operand order repaired and checked by reading, not compiled, not run on hardware
//
#include <hip/hip_runtime.h>
#include <hip/hip_bf16.h>

#define B_ 256
#define I_ 256
#define T_ 512
#define E_ 512
#define HROW 520   // padded LDS row (ushort units): 1040B stride -> conflict-free banks

typedef __attribute__((ext_vector_type(16))) __bf16 v16bf;
typedef __attribute__((ext_vector_type(8)))  float  v8f;
typedef __attribute__((ext_vector_type(8)))  unsigned short ushort8;

union V16 {
    v16bf v;
    unsigned short s[16];
    ushort8 h8[2];
};

// Native bf16 convert (v_cvt_bf16_f32 on gfx1250) instead of bit-trick RNE.
__device__ __forceinline__ unsigned short f32_to_bf16(float f) {
    __bf16 h = (__bf16)f;
    return __builtin_bit_cast(unsigned short, h);
}

__device__ __forceinline__ float bf16_to_f32(unsigned short s) {
    unsigned int u = ((unsigned int)s) << 16;
    return __builtin_bit_cast(float, u);
}

__device__ __forceinline__ float fast_tanh(float x) {
#if __has_builtin(__builtin_amdgcn_tanhf)
    return __builtin_amdgcn_tanhf(x);      // V_TANH_F32 on gfx1250
#else
    return tanhf(x);
#endif
}

// ---------------------------------------------------------------------------
// Kernel 0: convert weights f32 -> bf16 (one pass; both L2-resident afterwards)
// ---------------------------------------------------------------------------
__global__ void cvt_weights_kernel(const float* __restrict__ Wih,
                                   const float* __restrict__ Whh,
                                   unsigned short* __restrict__ Wih_bf,
                                   unsigned short* __restrict__ Whh_bf) {
    int i = blockIdx.x * blockDim.x + threadIdx.x;
    if (i < E_ * I_) Wih_bf[i] = f32_to_bf16(Wih[i]);
    if (i < E_ * E_) Whh_bf[i] = f32_to_bf16(Whh[i]);
}

// ---------------------------------------------------------------------------
// Kernel 1: xw[b,t,e] = sum_i x[b,i,t] * W_ih[e,i]   (bf16 WMMA, f32 accum)
// One wave computes a 16(t) x 64(e) tile; K-loop over I in chunks of 32.
// Memory-bound: x read once (134 MB), xw written once (128 MB bf16).
// ---------------------------------------------------------------------------
__global__ __launch_bounds__(256) void xw_gemm_kernel(
        const float* __restrict__ x,
        const unsigned short* __restrict__ Wih,     // bf16 [E][I]
        unsigned short* __restrict__ xw) {          // bf16 [B][T][E]
    const int lane = threadIdx.x & 31;
    const int wave = threadIdx.x >> 5;

    const long gw  = (long)blockIdx.x * 8 + wave;   // global wave id
    const int  e64 = (int)(gw % (E_ / 64));
    const int  tt  = (int)((gw / (E_ / 64)) % (T_ / 16));
    const int  b   = (int)(gw / ((E_ / 64) * (T_ / 16)));
    const int  t0  = tt * 16;
    const int  e0  = e64 * 64;

    const int  row    = lane & 15;
    const bool hi     = lane >= 16;
    const int  kbaseA = hi ? 8  : 0;   // A-matrix 16x32 bf16 layout
    const int  kbaseB = hi ? 16 : 0;   // contiguous k order (k = 16h + s), used for both A and B

    v8f c0 = {}, c1 = {}, c2 = {}, c3 = {};

    const float* xbase = x + (long)b * I_ * T_ + (t0 + row);     // A row = t
    const unsigned short* wbase = Wih + (long)(e0 + row) * I_;   // B col = e

    #pragma unroll
    for (int c = 0; c < I_ / 32; ++c) {
        const int k0 = c * 32;
        // A tile (x^T), per-lane gather + native cvt into contiguous k order (k = 16h + s)
        V16 a;
        #pragma unroll
        for (int h = 0; h < 16; ++h) {
            const int K = kbaseB + h;
            a.s[h] = f32_to_bf16(xbase[(long)(k0 + K) * T_]);
        }
        // B tiles (W_ih^T): K contiguous in memory -> 32B vector loads
        v16bf b0 = *(const v16bf*)(wbase + 0 * 16 * I_ + k0 + kbaseB);
        v16bf b1 = *(const v16bf*)(wbase + 1 * 16 * I_ + k0 + kbaseB);
        v16bf b2 = *(const v16bf*)(wbase + 2 * 16 * I_ + k0 + kbaseB);
        v16bf b3 = *(const v16bf*)(wbase + 3 * 16 * I_ + k0 + kbaseB);

        c0 = __builtin_amdgcn_wmma_f32_16x16x32_bf16(false, a.v, false, b0, (short)0, c0, false, false);
        c1 = __builtin_amdgcn_wmma_f32_16x16x32_bf16(false, a.v, false, b1, (short)0, c1, false, false);
        c2 = __builtin_amdgcn_wmma_f32_16x16x32_bf16(false, a.v, false, b2, (short)0, c2, false, false);
        c3 = __builtin_amdgcn_wmma_f32_16x16x32_bf16(false, a.v, false, b3, (short)0, c3, false, false);
    }

    // Store D as bf16: row t = t0 + r + (hi?8:0), col e = e0 + ntile*16 + row
    unsigned short* obase = xw + ((long)b * T_ + t0 + (hi ? 8 : 0)) * E_ + e0 + row;
    #pragma unroll
    for (int r = 0; r < 8; ++r) {
        obase[(long)r * E_ + 0]  = f32_to_bf16(c0[r]);
        obase[(long)r * E_ + 16] = f32_to_bf16(c1[r]);
        obase[(long)r * E_ + 32] = f32_to_bf16(c2[r]);
        obase[(long)r * E_ + 48] = f32_to_bf16(c3[r]);
    }
}

// ---------------------------------------------------------------------------
// Kernel 2: persistent recurrence. 16 blocks (one per 16-row batch tile),
// 512 threads = 16 waves; wave w owns e-columns [32w, 32w+32) with its
// W_hh^T slice held in 256 VGPRs. h double-buffered in LDS (bf16).
// xw loads are software-pipelined one step ahead so their global latency
// hides behind the 32-WMMA chain of the current step.
// ---------------------------------------------------------------------------
__global__ __launch_bounds__(512) void rnn_scan_kernel(
        const unsigned short* __restrict__ xw,     // bf16 [B][T][E]
        const unsigned short* __restrict__ Whh,    // bf16 [E][E]
        float* __restrict__ out) {                 // f32 [B][E]
    __shared__ unsigned short H[2][16][HROW];

    const int  lane   = threadIdx.x & 31;
    const int  wave   = threadIdx.x >> 5;      // 0..15
    const int  e0     = wave * 32;
    const int  b0     = blockIdx.x * 16;
    const int  row    = lane & 15;
    const bool hi     = lane >= 16;
    const int  kbaseA = hi ? 8  : 0;
    const int  kbaseB = hi ? 16 : 0;
    const int  r0     = hi ? 8  : 0;

    // Register-resident W_hh^T slice: B[k][n] = Whh[e0+nt*16+row][k]
    v16bf W0[16], W1[16];
    {
        const unsigned short* w0 = Whh + (long)(e0 + row) * E_;
        const unsigned short* w1 = Whh + (long)(e0 + 16 + row) * E_;
        #pragma unroll
        for (int c = 0; c < 16; ++c) {
            W0[c] = *(const v16bf*)(w0 + c * 32 + kbaseB);
            W1[c] = *(const v16bf*)(w1 + c * 32 + kbaseB);
        }
    }

    // h0 = 0
    for (int i = threadIdx.x; i < 16 * HROW; i += blockDim.x)
        (&H[0][0][0])[i] = 0;

    // Preload xw for t = 0 (base pointer excludes the t term)
    const unsigned short* xwrow = xw + (long)(b0 + r0) * T_ * E_ + e0 + row;
    unsigned short xc0[8], xc1[8];
    #pragma unroll
    for (int r = 0; r < 8; ++r) {
        xc0[r] = xwrow[(long)r * T_ * E_];
        xc1[r] = xwrow[(long)r * T_ * E_ + 16];
    }

    float h0f[8], h1f[8];
    __syncthreads();

    for (int t = 0; t < T_; ++t) {
        const unsigned short* cur = &H[t & 1][0][0];
        unsigned short*       nxt = &H[(t + 1) & 1][0][0];

        // Issue next timestep's xw loads now; latency overlaps the WMMA chain.
        const long tn = (t + 1 < T_) ? (t + 1) : t;
        unsigned short xn0[8], xn1[8];
        #pragma unroll
        for (int r = 0; r < 8; ++r) {
            xn0[r] = xwrow[(long)r * T_ * E_ + tn * E_];
            xn1[r] = xwrow[(long)r * T_ * E_ + tn * E_ + 16];
        }

        // C := h_prev @ W_hh^T  (16 chunks of K=32, accumulate from zero)
        v8f c0 = {}, c1 = {};
        #pragma unroll
        for (int c = 0; c < 16; ++c) {
            V16 a;
            a.h8[0] = *(const ushort8*)(cur + (long)row * HROW + c * 32 + kbaseB);
            a.h8[1] = *(const ushort8*)(cur + (long)row * HROW + c * 32 + 8 + kbaseB);
            c0 = __builtin_amdgcn_wmma_f32_16x16x32_bf16(false, a.v, false, W0[c], (short)0, c0, false, false);
            c1 = __builtin_amdgcn_wmma_f32_16x16x32_bf16(false, a.v, false, W1[c], (short)0, c1, false, false);
        }

        // h_new = tanh(C + xw_t); publish bf16 to next buffer
        #pragma unroll
        for (int r = 0; r < 8; ++r) {
            const float v0 = fast_tanh(c0[r] + bf16_to_f32(xc0[r]));
            const float v1 = fast_tanh(c1[r] + bf16_to_f32(xc1[r]));
            h0f[r] = v0;
            h1f[r] = v1;
            nxt[(long)(r0 + r) * HROW + e0 + row]      = f32_to_bf16(v0);
            nxt[(long)(r0 + r) * HROW + e0 + 16 + row] = f32_to_bf16(v1);
            xc0[r] = xn0[r];
            xc1[r] = xn1[r];
        }
        __syncthreads();
    }

    // Final h (full f32, pre-rounding) -> d_out
    #pragma unroll
    for (int r = 0; r < 8; ++r) {
        out[(long)(b0 + r0 + r) * E_ + e0 + row]      = h0f[r];
        out[(long)(b0 + r0 + r) * E_ + e0 + 16 + row] = h1f[r];
    }
}

// ---------------------------------------------------------------------------
extern "C" void kernel_launch(void* const* d_in, const int* in_sizes, int n_in,
                              void* d_out, int out_size, void* d_ws, size_t ws_size,
                              hipStream_t stream) {
    const float* x   = (const float*)d_in[0];   // [B, I, T]
    const float* Wih = (const float*)d_in[1];   // [E, I]
    const float* Whh = (const float*)d_in[2];   // [E, E]
    float* out = (float*)d_out;                 // [B, E]

    unsigned short* Whh_bf = (unsigned short*)d_ws;                              // 512 KB
    unsigned short* Wih_bf = (unsigned short*)((char*)d_ws + (512 << 10));       // 256 KB
    unsigned short* xw     = (unsigned short*)((char*)d_ws + (1 << 20));         // 128 MB

    // Kernel 0: weight conversion
    {
        int n = E_ * E_;
        cvt_weights_kernel<<<(n + 255) / 256, 256, 0, stream>>>(Wih, Whh, Wih_bf, Whh_bf);
    }
    // Kernel 1: input projection GEMM
    {
        long waves  = (long)B_ * (T_ / 16) * (E_ / 64);   // 65536
        int  blocks = (int)(waves / 8);                   // 8192
        xw_gemm_kernel<<<blocks, 256, 0, stream>>>(x, Wih_bf, xw);
    }
    // Kernel 2: recurrence
    {
        rnn_scan_kernel<<<B_ / 16, 512, 0, stream>>>(xw, Whh_bf, out);
    }
}
